// MelToWave_42339787604836
// MI455X (gfx1250) — compile-verified, matrix operand order checked by reading, not run on hardware
//
#include <hip/hip_runtime.h>
#include <hip/hip_bf16.h>
#include <math.h>

#define TB 256

typedef __attribute__((ext_vector_type(16))) _Float16 v16h;
typedef __attribute__((ext_vector_type(8)))  _Float16 v8h;
typedef __attribute__((ext_vector_type(8)))  float    v8f;

static constexpr int   kB    = 8;
static constexpr int   kT    = 2048;
static constexpr int   kF    = 513;
static constexpr int   kM    = 80;
static constexpr int   kNFFT = 1024;
static constexpr int   kHOP  = 256;
static constexpr int   kPAD  = 512;
static constexpr int   kL    = (kT - 1) * kHOP;   // 524032
static constexpr int   kLP   = kL + 2 * kPAD;     // 525056
static constexpr int   kKI   = 1056;              // istft K: [Sre;Sim] padded to 33*32
static constexpr int   kMS   = 1040;              // stft rows: 2*513 interleaved, pad to 65*16
static constexpr float kMOM  = 0.99f / 1.99f;
static constexpr float kTWOPI = 6.28318530717958647692f;

__device__ __forceinline__ float hannw(int n) {
  return 0.5f - 0.5f * cosf(kTWOPI * (float)n * (1.0f / 1024.0f));
}

// ---------------------------------------------------------------------------
// Fragment helpers. 16-bit A/B 16x32 layout: lanes 0-15 -> K {0..7,16..23},
// lanes 16-31 -> K {8..15,24..31}; two 16B loads per fragment.
// ---------------------------------------------------------------------------
__device__ __forceinline__ v16h load_frag16x32(const _Float16* base, int kb) {
  union { v16h v; v8h h[2]; } f;
  f.h[0] = *(const v8h*)(base + kb);
  f.h[1] = *(const v8h*)(base + kb + 16);
  return f.v;
}

// Async-stage a 16-row x Kp panel of A (row-major) into LDS, mirrored layout.
// Uses GLOBAL_LOAD_ASYNC_TO_LDS_B128 (ASYNCcnt) + s_wait_asynccnt.
__device__ __forceinline__ void stage_A_async(const _Float16* Ag, _Float16* sA,
                                              int Kp) {
  const int chunks = (16 * Kp) >> 3;  // 16-byte chunks
  unsigned ldsBase = (unsigned)(uintptr_t)sA;
  for (int j = threadIdx.x; j < chunks; j += blockDim.x) {
    const _Float16* g = Ag + (size_t)j * 8;
    unsigned l = ldsBase + (unsigned)j * 16u;
    asm volatile("global_load_async_to_lds_b128 %0, %1, off"
                 :: "v"(l), "v"(g) : "memory");
  }
  asm volatile("s_wait_asynccnt 0x0" ::: "memory");
  __syncthreads();
}

// Software-pipelined K-loop body shared by both GEMMs: fragments for step
// k0+32 are issued before the WMMA consuming step k0, hiding load latency.
__device__ __forceinline__ v8f gemm_k_loop(const _Float16* As,
                                           const _Float16* Bp,
                                           int Kp, int kb) {
  v8f acc = {};
  v16h af = load_frag16x32(As, kb);
  v16h bf = load_frag16x32(Bp, kb);
  int k0 = 0;
  for (; k0 + 32 < Kp; k0 += 32) {
    if (k0 + 160 < Kp) __builtin_prefetch(Bp + k0 + 160, 0, 3);
    v16h afn = load_frag16x32(As + k0 + 32, kb);
    v16h bfn = load_frag16x32(Bp + k0 + 32, kb);
    acc = __builtin_amdgcn_wmma_f32_16x16x32_f16(false, af, false, bf,
                                                 (short)0, acc, false, false);
    af = afn;
    bf = bfn;
  }
  acc = __builtin_amdgcn_wmma_f32_16x16x32_f16(false, af, false, bf,
                                               (short)0, acc, false, false);
  return acc;
}

// ---------------------------------------------------------------------------
// ISTFT GEMM: frames(b,t,1024) f16 = IC(1024x1056) @ Ssig(b; col t, k).
// Block: 1 M-tile x 8 N-tiles; A panel shared via LDS (async staged).
// ---------------------------------------------------------------------------
__global__ void __launch_bounds__(256) gemm_istft_wmma(
    const _Float16* __restrict__ A, const _Float16* __restrict__ Bm,
    _Float16* __restrict__ C, int Mp, int Kp, int Nn) {
  __shared__ _Float16 sA[16 * kKI];
  const int lane = threadIdx.x & 31;
  const int wid  = threadIdx.x >> 5;
  const int mtiles = Mp >> 4;
  const int ntg    = Nn >> 7;                 // 128 columns per block
  const int mt = blockIdx.x % mtiles;
  const int rest = blockIdx.x / mtiles;
  const int ng = rest % ntg;
  const int b  = rest / ntg;
  const int nt = (ng << 3) + wid;
  const int l15 = lane & 15;
  const int kb  = (lane & 16) >> 1;

  stage_A_async(A + (size_t)(mt << 4) * Kp, sA, Kp);

  const _Float16* As = sA + (size_t)l15 * Kp;
  const _Float16* Bp = Bm + ((size_t)b * Nn + (nt << 4) + l15) * Kp;

  v8f acc = gemm_k_loop(As, Bp, Kp, kb);

  // C VGPR r holds (M = m0 + r, N = lane&15), m0 base = 8*(lane/16)
  _Float16* Cp = C + ((size_t)b * Nn + (nt << 4) + l15) * Mp +
                 (mt << 4) + ((lane >> 4) << 3);
  v8h hv;
#pragma unroll
  for (int r = 0; r < 8; ++r) hv[r] = (_Float16)acc[r];
  *(v8h*)Cp = hv;
}

// ---------------------------------------------------------------------------
// STFT GEMM + fused Griffin-Lim phase update.
// FC rows interleaved: row 2f = cos(2pi f n/N), row 2f+1 = -sin(...), so each
// lane's accumulators hold 4 (re,im) pairs; momentum update + normalize happen
// in registers and no rebuilt-spectrum buffer is ever written.
// ---------------------------------------------------------------------------
__global__ void __launch_bounds__(256) gemm_stft_update_wmma(
    const _Float16* __restrict__ A, const _Float16* __restrict__ Bm,
    float* __restrict__ aR, float* __restrict__ aI,
    float* __restrict__ tR, float* __restrict__ tI,
    int Mp, int Kp, int Nn) {
  __shared__ _Float16 sA[16 * kNFFT];
  const int lane = threadIdx.x & 31;
  const int wid  = threadIdx.x >> 5;
  const int mtiles = Mp >> 4;
  const int ntg    = Nn >> 7;
  const int mt = blockIdx.x % mtiles;
  const int rest = blockIdx.x / mtiles;
  const int ng = rest % ntg;
  const int b  = rest / ntg;
  const int nt = (ng << 3) + wid;
  const int l15 = lane & 15;
  const int kb  = (lane & 16) >> 1;

  stage_A_async(A + (size_t)(mt << 4) * Kp, sA, Kp);

  const _Float16* As = sA + (size_t)l15 * Kp;
  const _Float16* Bp = Bm + ((size_t)b * Nn + (nt << 4) + l15) * Kp;

  v8f acc = gemm_k_loop(As, Bp, Kp, kb);

  const int t  = (nt << 4) + l15;
  const int m0 = (mt << 4) + ((lane >> 4) << 3);
  const int f0 = m0 >> 1;
#pragma unroll
  for (int j = 0; j < 4; ++j) {
    int f = f0 + j;
    if (f < kF) {
      size_t ix = ((size_t)b * kF + f) * kT + t;
      float re = acc[2 * j], im = acc[2 * j + 1];
      float ar = re - kMOM * tR[ix];
      float ai = im - kMOM * tI[ix];
      float inv = 1.0f / (sqrtf(ar * ar + ai * ai) + 1e-16f);
      aR[ix] = ar * inv;
      aI[ix] = ai * inv;
      tR[ix] = re;
      tI[ix] = im;
    }
  }
}

// ---------------------------------------------------------------------------
// Setup kernels
// ---------------------------------------------------------------------------
__global__ void k_mel(const float* __restrict__ lm, float* __restrict__ mel, int n) {
  int i = blockIdx.x * blockDim.x + threadIdx.x;
  if (i >= n) return;
  float v = expm1f(lm[i]);
  mel[i] = v > 0.f ? v : 0.f;
}

__global__ void k_gram(const float* __restrict__ fb, float* __restrict__ G) {
  int i = blockIdx.x * blockDim.x + threadIdx.x;
  if (i >= kM * kM) return;
  int r = i / kM, c = i % kM;
  float s = 0.f;
  for (int f = 0; f < kF; f++) s += fb[f * kM + r] * fb[f * kM + c];
  G[i] = s;
}

__global__ void __launch_bounds__(256) k_inv(const float* __restrict__ G,
                                             float* __restrict__ Gi) {
  __shared__ float aug[kM][2 * kM + 1];
  __shared__ float colp[kM];
  __shared__ float pivinv;
  const int tid = threadIdx.x, nt = blockDim.x;
  for (int i = tid; i < kM * kM; i += nt) {
    int r = i / kM, c = i % kM;
    aug[r][c] = G[i];
    aug[r][kM + c] = (r == c) ? 1.f : 0.f;
  }
  __syncthreads();
  for (int p = 0; p < kM; p++) {
    if (tid == 0) pivinv = 1.0f / aug[p][p];
    __syncthreads();
    for (int c = tid; c < 2 * kM; c += nt) aug[p][c] *= pivinv;
    __syncthreads();
    for (int r = tid; r < kM; r += nt) colp[r] = aug[r][p];
    __syncthreads();
    for (int i = tid; i < kM * 2 * kM; i += nt) {
      int r = i / (2 * kM), c = i % (2 * kM);
      if (r != p) aug[r][c] -= colp[r] * aug[p][c];
    }
    __syncthreads();
  }
  for (int i = tid; i < kM * kM; i += nt) {
    int r = i / kM, c = i % kM;
    Gi[i] = aug[r][kM + c];
  }
}

__global__ void k_pmat(const float* __restrict__ fb, const float* __restrict__ Gi,
                       float* __restrict__ P) {
  int i = blockIdx.x * blockDim.x + threadIdx.x;
  if (i >= kF * kM) return;
  int f = i / kM, m = i % kM;
  float s = 0.f;
  for (int j = 0; j < kM; j++) s += fb[f * kM + j] * Gi[j * kM + m];
  P[i] = s;
}

__global__ void k_spec(const float* __restrict__ P, const float* __restrict__ mel,
                       float* __restrict__ spec) {
  int i = blockIdx.x * blockDim.x + threadIdx.x;
  if (i >= kB * kF * kT) return;
  int t = i % kT, f = (i / kT) % kF, b = i / (kF * kT);
  const float* Pp = P + f * kM;
  const float* mp = mel + (size_t)b * kM * kT + t;
  float s = 0.f;
  for (int m = 0; m < kM; m++) s += Pp[m] * mp[(size_t)m * kT];
  spec[i] = s > 0.f ? s : 0.f;
}

// ISTFT combine matrix IC (1024 x 1056): window*(1/N)*Hermitian-fold
__global__ void k_build_ic(_Float16* __restrict__ IC) {
  int i = blockIdx.x * blockDim.x + threadIdx.x;
  if (i >= kNFFT * kKI) return;
  int n = i / kKI, k = i % kKI;
  float wn = hannw(n), v = 0.f;
  if (k < kF) {
    float c = (k == 0 || k == 512) ? 1.f : 2.f;
    int ph = (n * k) & 1023;
    v = wn * (1.f / 1024.f) * c * cosf(kTWOPI * (float)ph * (1.f / 1024.f));
  } else if (k < 2 * kF) {
    int kk = k - kF;
    float c = (kk == 0 || kk == 512) ? 1.f : 2.f;
    int ph = (n * kk) & 1023;
    v = -wn * (1.f / 1024.f) * c * sinf(kTWOPI * (float)ph * (1.f / 1024.f));
  }
  IC[i] = (_Float16)v;
}

// Forward DFT matrix FC (1040 x 1024), rows INTERLEAVED: row 2f = cos,
// row 2f+1 = -sin. Rows >= 1026 are zero padding.
__global__ void k_build_fc(_Float16* __restrict__ FC) {
  int i = blockIdx.x * blockDim.x + threadIdx.x;
  if (i >= kMS * kNFFT) return;
  int r = i / kNFFT, n = i % kNFFT;
  float v = 0.f;
  if (r < 2 * kF) {
    int f = r >> 1;
    int ph = (f * n) & 1023;
    float c = kTWOPI * (float)ph * (1.f / 1024.f);
    v = (r & 1) ? -sinf(c) : cosf(c);
  }
  FC[i] = (_Float16)v;
}

__global__ void k_env(float* __restrict__ env) {
  int p = blockIdx.x * blockDim.x + threadIdx.x;
  if (p >= kLP) return;
  int tlo = (p >= kNFFT) ? ((p - kNFFT) >> 8) + 1 : 0;
  int thi = p >> 8;
  if (thi > kT - 1) thi = kT - 1;
  float e = 0.f;
  for (int t = tlo; t <= thi; t++) {
    float wv = hannw(p - (t << 8));
    e += wv * wv;
  }
  env[p] = (e > 1e-11f) ? e : 1.0f;
}

__global__ void k_init(const float* __restrict__ ai, float* __restrict__ aR,
                       float* __restrict__ aI, float* __restrict__ tR,
                       float* __restrict__ tI) {
  int i = blockIdx.x * blockDim.x + threadIdx.x;
  if (i >= kB * kF * kT) return;
  aR[i] = ai[2 * (size_t)i];
  aI[i] = ai[2 * (size_t)i + 1];
  tR[i] = 0.f;
  tI[i] = 0.f;
}

// ---------------------------------------------------------------------------
// Per-iteration kernels
// ---------------------------------------------------------------------------
__global__ void k_pack(const float* __restrict__ spec, const float* __restrict__ aR,
                       const float* __restrict__ aI, _Float16* __restrict__ S) {
  int i = blockIdx.x * blockDim.x + threadIdx.x;
  if (i >= kB * kT * kKI) return;
  int k = i % kKI;
  int bt = i / kKI;
  int t = bt % kT, b = bt / kT;
  float v = 0.f;
  if (k < kF) {
    size_t s = ((size_t)b * kF + k) * kT + t;
    v = spec[s] * aR[s];
  } else if (k < 2 * kF) {
    size_t s = ((size_t)b * kF + (k - kF)) * kT + t;
    v = spec[s] * aI[s];
  }
  S[i] = (_Float16)v;
}

// Overlap-add + envelope normalize + center crop (pure gather, <=4 frames/sample)
__global__ void k_ola(const _Float16* __restrict__ fr, const float* __restrict__ env,
                      float* __restrict__ y) {
  int i = blockIdx.x * blockDim.x + threadIdx.x;
  if (i >= kB * kL) return;
  int p = kPAD + (i % kL);
  int b = i / kL;
  int tlo = (p >= kNFFT) ? ((p - kNFFT) >> 8) + 1 : 0;
  int thi = p >> 8;
  if (thi > kT - 1) thi = kT - 1;
  float s = 0.f;
  for (int t = tlo; t <= thi; t++)
    s += (float)fr[(((size_t)b * kT + t) << 10) + (p - (t << 8))];
  y[i] = s / env[p];
}

// Reflect-pad framing + window -> f16 frames (B,T,1024) for GEMM B operand
__global__ void k_frame(const float* __restrict__ yc, _Float16* __restrict__ out) {
  int i = blockIdx.x * blockDim.x + threadIdx.x;
  if (i >= kB * kT * kNFFT) return;
  int k = i & 1023;
  int t = (i >> 10) % kT;
  int b = i / (kT * kNFFT);
  int x = t * kHOP + k - kPAD;
  if (x < 0) x = -x;
  if (x >= kL) x = 2 * kL - 2 - x;
  out[i] = (_Float16)(yc[(size_t)b * kL + x] * hannw(k));
}

// ---------------------------------------------------------------------------
extern "C" void kernel_launch(void* const* d_in, const int* in_sizes, int n_in,
                              void* d_out, int out_size, void* d_ws, size_t ws_size,
                              hipStream_t stream) {
  (void)in_sizes; (void)n_in; (void)out_size; (void)ws_size;
  const float* logmel = (const float*)d_in[0];
  const float* fb     = (const float*)d_in[1];
  const float* ainit  = (const float*)d_in[2];  // complex64 interleaved

  char* w = (char*)d_ws;
  auto alloc = [&](size_t bytes) -> void* {
    void* p = (void*)w;
    w += (bytes + 255) & ~(size_t)255;
    return p;
  };
  float*    G    = (float*)alloc((size_t)kM * kM * 4);
  float*    Gi   = (float*)alloc((size_t)kM * kM * 4);
  float*    P    = (float*)alloc((size_t)kF * kM * 4);
  float*    mel  = (float*)alloc((size_t)kB * kM * kT * 4);
  float*    spec = (float*)alloc((size_t)kB * kF * kT * 4);
  float*    aR   = (float*)alloc((size_t)kB * kF * kT * 4);
  float*    aI   = (float*)alloc((size_t)kB * kF * kT * 4);
  float*    tR   = (float*)alloc((size_t)kB * kF * kT * 4);
  float*    tI   = (float*)alloc((size_t)kB * kF * kT * 4);
  _Float16* IC   = (_Float16*)alloc((size_t)kNFFT * kKI * 2);
  _Float16* FC   = (_Float16*)alloc((size_t)kMS * kNFFT * 2);
  _Float16* Ssig = (_Float16*)alloc((size_t)kB * kT * kKI * 2);
  _Float16* frO  = (_Float16*)alloc((size_t)kB * kT * kNFFT * 2);
  _Float16* fr2  = (_Float16*)alloc((size_t)kB * kT * kNFFT * 2);
  float*    yc   = (float*)alloc((size_t)kB * kL * 4);
  float*    env  = (float*)alloc((size_t)kLP * 4);

  auto nb = [](long long n) { return (unsigned)((n + TB - 1) / TB); };
  const int NBFT = kB * kF * kT;

  // ---- setup ----
  k_mel<<<nb((long long)kB * kM * kT), TB, 0, stream>>>(logmel, mel, kB * kM * kT);
  k_gram<<<nb(kM * kM), TB, 0, stream>>>(fb, G);
  k_inv<<<1, 256, 0, stream>>>(G, Gi);
  k_pmat<<<nb(kF * kM), TB, 0, stream>>>(fb, Gi, P);
  k_spec<<<nb(NBFT), TB, 0, stream>>>(P, mel, spec);
  k_build_ic<<<nb((long long)kNFFT * kKI), TB, 0, stream>>>(IC);
  k_build_fc<<<nb((long long)kMS * kNFFT), TB, 0, stream>>>(FC);
  k_env<<<nb(kLP), TB, 0, stream>>>(env);
  k_init<<<nb(NBFT), TB, 0, stream>>>(ainit, aR, aI, tR, tI);

  // GEMM grids: block = 1 M-tile x 8 N-tiles (128 cols), 8 waves/block
  const unsigned gIstft = (unsigned)((kNFFT / 16) * (kT / 128) * kB);  // 8192
  const unsigned gStft  = (unsigned)((kMS / 16) * (kT / 128) * kB);    // 8320

  // ---- Griffin-Lim iterations ----
  for (int it = 0; it < 60; ++it) {
    k_pack<<<nb((long long)kB * kT * kKI), TB, 0, stream>>>(spec, aR, aI, Ssig);
    gemm_istft_wmma<<<gIstft, 256, 0, stream>>>(IC, Ssig, frO, kNFFT, kKI, kT);
    k_ola<<<nb((long long)kB * kL), TB, 0, stream>>>(frO, env, yc);
    k_frame<<<nb((long long)kB * kT * kNFFT), TB, 0, stream>>>(yc, fr2);
    gemm_stft_update_wmma<<<gStft, 256, 0, stream>>>(FC, fr2, aR, aI, tR, tI,
                                                     kMS, kNFFT, kT);
  }

  // ---- final ISTFT -> d_out ----
  k_pack<<<nb((long long)kB * kT * kKI), TB, 0, stream>>>(spec, aR, aI, Ssig);
  gemm_istft_wmma<<<gIstft, 256, 0, stream>>>(IC, Ssig, frO, kNFFT, kKI, kT);
  k_ola<<<nb((long long)kB * kL), TB, 0, stream>>>(frO, env, (float*)d_out);
}
